// HybridQCNNQLSTM_65481071408542
// MI455X (gfx1250) — matrix operand order repaired and checked by reading, not compiled, not run on hardware
//
#include <hip/hip_runtime.h>
#include <hip/hip_bf16.h>

typedef __attribute__((ext_vector_type(16))) _Float16 v16h;
typedef __attribute__((ext_vector_type(8)))  float    v8f;

#define B_    256
#define S_    1024
#define H_    256
#define GDIM  257      // 1 + H
#define NBLK  16       // persistent LSTM workgroups

__device__ __forceinline__ float sigmf(float x) { return 1.0f / (1.0f + __expf(-x)); }

// ---------------------------------------------------------------------------
// Init: zero h ping buffer + barrier state (deterministic per launch)
// ---------------------------------------------------------------------------
__global__ void init_kernel(unsigned int* __restrict__ h0_u32, int* __restrict__ bar) {
    int i = blockIdx.x * blockDim.x + threadIdx.x;
    if (i < (B_ * H_ / 2)) h0_u32[i] = 0u;   // B_*H_ halfs as uints
    if (i < 2) bar[i] = 0;
}

// ---------------------------------------------------------------------------
// CNN-MLP per (b,s) element: 1->16->16->12->8->4->4->1, tanh except last.
// Writes cnn_features (B,S) to d_out and time-major (S,B) copy to ws.
// ---------------------------------------------------------------------------
__global__ void mlp_kernel(const float* __restrict__ x,
                           const float* __restrict__ w1, const float* __restrict__ b1,
                           const float* __restrict__ w2, const float* __restrict__ b2,
                           const float* __restrict__ w3, const float* __restrict__ b3,
                           const float* __restrict__ w4, const float* __restrict__ b4,
                           const float* __restrict__ w5, const float* __restrict__ b5,
                           const float* __restrict__ w6, const float* __restrict__ b6,
                           const float* __restrict__ w7, const float* __restrict__ b7,
                           float* __restrict__ feat_out, float* __restrict__ xseq) {
    __shared__ float s_w1[16], s_b1[16], s_w2[256], s_b2[16], s_w3[192], s_b3[12],
                     s_w4[96], s_b4[8], s_w5[32], s_b5[4], s_w6[16], s_b6[4],
                     s_w7[4], s_b7[1];
    const int t = threadIdx.x;
    if (t < 16)  s_w1[t] = w1[t];
    if (t < 16)  s_b1[t] = b1[t];
    s_w2[t] = w2[t];
    if (t < 16)  s_b2[t] = b2[t];
    if (t < 192) s_w3[t] = w3[t];
    if (t < 12)  s_b3[t] = b3[t];
    if (t < 96)  s_w4[t] = w4[t];
    if (t < 8)   s_b4[t] = b4[t];
    if (t < 32)  s_w5[t] = w5[t];
    if (t < 4)   s_b5[t] = b5[t];
    if (t < 16)  s_w6[t] = w6[t];
    if (t < 4)   s_b6[t] = b6[t];
    if (t < 4)   s_w7[t] = w7[t];
    if (t < 1)   s_b7[t] = b7[t];
    __syncthreads();

    const int idx = blockIdx.x * 256 + t;          // 0 .. B*S-1
    const float xin = x[idx];
    float va[16], vb[16];
#pragma unroll
    for (int o = 0; o < 16; ++o) va[o] = tanhf(s_w1[o] * xin + s_b1[o]);
#pragma unroll
    for (int o = 0; o < 16; ++o) {
        float s = s_b2[o];
#pragma unroll
        for (int i = 0; i < 16; ++i) s += va[i] * s_w2[o * 16 + i];
        vb[o] = tanhf(s);
    }
#pragma unroll
    for (int o = 0; o < 12; ++o) {
        float s = s_b3[o];
#pragma unroll
        for (int i = 0; i < 16; ++i) s += vb[i] * s_w3[o * 16 + i];
        va[o] = tanhf(s);
    }
#pragma unroll
    for (int o = 0; o < 8; ++o) {
        float s = s_b4[o];
#pragma unroll
        for (int i = 0; i < 12; ++i) s += va[i] * s_w4[o * 12 + i];
        vb[o] = tanhf(s);
    }
#pragma unroll
    for (int o = 0; o < 4; ++o) {
        float s = s_b5[o];
#pragma unroll
        for (int i = 0; i < 8; ++i) s += vb[i] * s_w5[o * 8 + i];
        va[o] = tanhf(s);
    }
#pragma unroll
    for (int o = 0; o < 4; ++o) {
        float s = s_b6[o];
#pragma unroll
        for (int i = 0; i < 4; ++i) s += va[i] * s_w6[o * 4 + i];
        vb[o] = tanhf(s);
    }
    float f = s_b7[0];
#pragma unroll
    for (int i = 0; i < 4; ++i) f += vb[i] * s_w7[i];

    const int b = idx >> 10;          // / S_
    const int s = idx & (S_ - 1);     // % S_
    feat_out[idx] = f;                // (B,S) order for output
    xseq[(size_t)s * B_ + b] = f;     // time-major for LSTM
}

// ---------------------------------------------------------------------------
// Persistent WMMA LSTM. 16 blocks x 256 threads (8 wave32).
// Block bk owns hidden units [bk*16, bk*16+16); per step computes the
// (256 x 64) gate pre-activation slab via v_wmma_f32_16x16x32_f16 with
// recurrent weights held in registers as f16 B-fragments.
// ---------------------------------------------------------------------------
__global__ void __launch_bounds__(256, 1)
lstm_kernel(const float* __restrict__ xseq,
            const float* __restrict__ wf, const float* __restrict__ bf,
            const float* __restrict__ wi, const float* __restrict__ bi,
            const float* __restrict__ wg, const float* __restrict__ bg,
            const float* __restrict__ wo, const float* __restrict__ bo,
            const float* __restrict__ whead, const float* __restrict__ bhead,
            _Float16* __restrict__ hbuf0, _Float16* __restrict__ hbuf1,
            int* __restrict__ bar, float* __restrict__ logits) {
    const int tid   = threadIdx.x;
    const int lane  = tid & 31;
    const int wave  = tid >> 5;
    const int n     = lane & 15;      // output column within tile (and M row for A)
    const int khalf = lane >> 4;      // which K half this lane carries
    const int bk    = blockIdx.x;
    const int ub    = bk * 16;        // hidden-unit base
    const int j     = ub + n;         // weight row (hidden unit) for this lane
    const int rb    = wave * 32;      // batch-row base for this wave

    const float* W[4]  = {wf, wi, wg, wo};
    const float* Bs[4] = {bf, bi, bg, bo};

    // --- Preload recurrent weights as B-fragments (32x16 f16 per K-chunk).
    // B layout: lane = column (n), lanes 0-15 carry K=0..15, lanes 16-31 K=16..31,
    // i.e. 16 contiguous K values per lane -> read W[j][1 + kc*32 + khalf*16 + t].
    v16h Bfrag[8][4];
#pragma unroll
    for (int kc = 0; kc < 8; ++kc) {
#pragma unroll
        for (int gi = 0; gi < 4; ++gi) {
            const float* src = W[gi] + (size_t)j * GDIM + 1 + kc * 32 + khalf * 16;
            v16h bb;
#pragma unroll
            for (int tt = 0; tt < 16; ++tt) bb[tt] = (_Float16)src[tt];
            Bfrag[kc][gi] = bb;
        }
    }
    float w0[4], bia[4];
#pragma unroll
    for (int gi = 0; gi < 4; ++gi) {
        w0[gi]  = W[gi][(size_t)j * GDIM];   // x_t column of the weight
        bia[gi] = Bs[gi][j];
    }

    // Cell state lives in registers: layout matches the D-tile (row,unit) map.
    float cst[2][8];
#pragma unroll
    for (int R = 0; R < 2; ++R)
#pragma unroll
        for (int r = 0; r < 8; ++r) cst[R][r] = 0.0f;

    for (int t = 0; t < S_; ++t) {
        const _Float16* hin  = (t & 1) ? hbuf1 : hbuf0;
        _Float16*       hout = (t & 1) ? hbuf0 : hbuf1;

        v8f acc[2][4] = {};

#pragma unroll
        for (int kc = 0; kc < 8; ++kc) {
            // A-fragments (16x32 f16): lane m = n; 16 contiguous K at
            // K = kc*32 + khalf*16 (matches B-fragment K order).
            v16h a0, a1;
            {
                const _Float16* p = hin + (size_t)(rb + n) * H_ + kc * 32 + khalf * 16;
                ((uint4*)&a0)[0] = *(const uint4*)p;
                ((uint4*)&a0)[1] = *(const uint4*)(p + 8);
            }
            {
                const _Float16* p = hin + (size_t)(rb + 16 + n) * H_ + kc * 32 + khalf * 16;
                ((uint4*)&a1)[0] = *(const uint4*)p;
                ((uint4*)&a1)[1] = *(const uint4*)(p + 8);
            }
#pragma unroll
            for (int gi = 0; gi < 4; ++gi) {
                acc[0][gi] = __builtin_amdgcn_wmma_f32_16x16x32_f16(
                    false, a0, false, Bfrag[kc][gi], (short)0, acc[0][gi], false, false);
                acc[1][gi] = __builtin_amdgcn_wmma_f32_16x16x32_f16(
                    false, a1, false, Bfrag[kc][gi], (short)0, acc[1][gi], false, false);
            }
        }

        // Gate math: D-tile VGPR r, lanes 0-15 -> M=r, lanes 16-31 -> M=r+8; N = n.
        const float* xs = xseq + (size_t)t * B_;
#pragma unroll
        for (int R = 0; R < 2; ++R) {
#pragma unroll
            for (int r = 0; r < 8; ++r) {
                const int row = rb + R * 16 + r + khalf * 8;  // batch index
                const float xt = xs[row];
                const float fp = acc[R][0][r] + xt * w0[0] + bia[0];
                const float ip = acc[R][1][r] + xt * w0[1] + bia[1];
                const float gp = acc[R][2][r] + xt * w0[2] + bia[2];
                const float op = acc[R][3][r] + xt * w0[3] + bia[3];
                const float cc = sigmf(fp) * cst[R][r] + sigmf(ip) * tanhf(gp);
                cst[R][r] = cc;
                const float hn = sigmf(op) * tanhf(cc);
                hout[(size_t)row * H_ + j] = (_Float16)hn;
            }
        }

        // ---- grid barrier across the 16 persistent blocks ----
        __syncthreads();
        if (tid == 0) {
            __threadfence();  // make h stores device-visible
            int g = __hip_atomic_load(&bar[1], __ATOMIC_ACQUIRE, __HIP_MEMORY_SCOPE_AGENT);
            int a = __hip_atomic_fetch_add(&bar[0], 1, __ATOMIC_ACQ_REL, __HIP_MEMORY_SCOPE_AGENT);
            if (a == NBLK - 1) {
                __hip_atomic_store(&bar[0], 0, __ATOMIC_RELAXED, __HIP_MEMORY_SCOPE_AGENT);
                __hip_atomic_fetch_add(&bar[1], 1, __ATOMIC_RELEASE, __HIP_MEMORY_SCOPE_AGENT);
            } else {
                while (__hip_atomic_load(&bar[1], __ATOMIC_ACQUIRE, __HIP_MEMORY_SCOPE_AGENT) == g)
                    __builtin_amdgcn_s_sleep(1);
            }
        }
        __syncthreads();
        __threadfence();  // invalidate near caches before reading peers' h
    }

    // Head: logits[b] = h_final[b] . w_head + b_head   (final h is in hbuf0)
    if (bk == 0) {
        const int b = tid;
        const _Float16* hr = hbuf0 + (size_t)b * H_;
        float s = bhead[0];
#pragma unroll 8
        for (int k = 0; k < H_; ++k) s += (float)hr[k] * whead[k];
        logits[b] = s;
    }
}

// ---------------------------------------------------------------------------
extern "C" void kernel_launch(void* const* d_in, const int* in_sizes, int n_in,
                              void* d_out, int out_size, void* d_ws, size_t ws_size,
                              hipStream_t stream) {
    const float* x  = (const float*)d_in[0];
    const float* w1 = (const float*)d_in[1];  const float* b1 = (const float*)d_in[2];
    const float* w2 = (const float*)d_in[3];  const float* b2 = (const float*)d_in[4];
    const float* w3 = (const float*)d_in[5];  const float* b3 = (const float*)d_in[6];
    const float* w4 = (const float*)d_in[7];  const float* b4 = (const float*)d_in[8];
    const float* w5 = (const float*)d_in[9];  const float* b5 = (const float*)d_in[10];
    const float* w6 = (const float*)d_in[11]; const float* b6 = (const float*)d_in[12];
    const float* w7 = (const float*)d_in[13]; const float* b7 = (const float*)d_in[14];
    const float* wf = (const float*)d_in[15]; const float* bf = (const float*)d_in[16];
    const float* wi = (const float*)d_in[17]; const float* bi = (const float*)d_in[18];
    const float* wg = (const float*)d_in[19]; const float* bg = (const float*)d_in[20];
    const float* wo = (const float*)d_in[21]; const float* bo = (const float*)d_in[22];
    const float* wh = (const float*)d_in[23]; const float* bh = (const float*)d_in[24];

    char* ws = (char*)d_ws;
    float*    xseq = (float*)ws;                                   // S*B f32 = 1 MB
    _Float16* h0   = (_Float16*)(ws + (size_t)S_ * B_ * sizeof(float));
    _Float16* h1   = h0 + (size_t)B_ * H_;                         // 2 x 128 KB f16
    int*      bar  = (int*)(h1 + (size_t)B_ * H_);                 // 2 ints

    float* feat   = (float*)d_out;                 // (B,S)
    float* logits = feat + (size_t)B_ * S_;        // (B,1)

    init_kernel<<<(B_ * H_ / 2 + 255) / 256, 256, 0, stream>>>((unsigned int*)h0, bar);
    mlp_kernel<<<(B_ * S_) / 256, 256, 0, stream>>>(x, w1, b1, w2, b2, w3, b3, w4, b4,
                                                    w5, b5, w6, b6, w7, b7, feat, xseq);
    lstm_kernel<<<NBLK, 256, 0, stream>>>(xseq, wf, bf, wi, bi, wg, bg, wo, bo,
                                          wh, bh, h0, h1, bar, logits);
}
